// BiLSTM_CRF_50259707298144
// MI455X (gfx1250) — matrix operand order repaired and checked by reading, not compiled, not run on hardware
//
#include <hip/hip_runtime.h>
#include <hip/hip_bf16.h>

// ---------------- problem constants ----------------
#define Lc    4096
#define EMBc  512
#define Hc    512
#define G4    2048        // 4*H gate rows
#define TAGSc 8
#define STARTc 6
#define STOPc  7
#define NEGc  (-10000.0f)

typedef __attribute__((ext_vector_type(16))) __bf16 v16bf;
typedef __attribute__((ext_vector_type(8)))  __bf16 v8bf;
typedef __attribute__((ext_vector_type(8)))  float  v8f;

union Frag16 {            // 16 bf16 elements = A or B fragment of wmma_f32_16x16x32_bf16
    v8bf  h[2];
    v16bf v;
};

struct BFrags {           // the four B fragments of one k-step (4 N-tiles)
    Frag16 b0, b1, b2, b3;
};

__device__ __forceinline__ unsigned short f32_to_bf16_rne(float f) {
    unsigned u = __float_as_uint(f);
    unsigned r = u + 0x7FFFu + ((u >> 16) & 1u);   // round-to-nearest-even
    return (unsigned short)(r >> 16);
}

__device__ __forceinline__ float fsigmoid(float x) {
    return 1.0f / (1.0f + __expf(-x));
}
__device__ __forceinline__ float ftanh(float x) {
    float e2 = __expf(2.0f * x);
    return (e2 - 1.0f) / (e2 + 1.0f);
}

// B fragment: col n fixed per lane, half 0 -> K 0..7 & 16..23 ; half 1 -> K 8..15 & 24..31
__device__ __forceinline__ void load_bfrag(Frag16& fb, const unsigned short* Brow,
                                           int k0, int half) {
    fb.h[0] = *(const v8bf*)(Brow + k0 + half * 8);
    fb.h[1] = *(const v8bf*)(Brow + k0 + half * 8 + 16);
}

__device__ __forceinline__ void load_b4(BFrags& s,
                                        const unsigned short* B0, const unsigned short* B1,
                                        const unsigned short* B2, const unsigned short* B3,
                                        int k0, int half) {
    load_bfrag(s.b0, B0, k0, half);
    load_bfrag(s.b1, B1, k0, half);
    load_bfrag(s.b2, B2, k0, half);
    load_bfrag(s.b3, B3, k0, half);
}

// ---------------- K0: gather embeddings + convert weights to bf16 ----------------
// emb16:  [L][EMB]            (gathered rows of wembed)
// wih16:  [2][4H][EMB]        W row-major == B-matrix column-major (contiguous-K per lane)
// whh16:  [2][4H][H]
__global__ void prep_kernel(const int* __restrict__ sent, const float* __restrict__ wembed,
                            const float* __restrict__ wihf, const float* __restrict__ wihb,
                            const float* __restrict__ whhf, const float* __restrict__ whhb,
                            unsigned short* __restrict__ emb16,
                            unsigned short* __restrict__ wih16,
                            unsigned short* __restrict__ whh16) {
    const long long NE = (long long)Lc * EMBc;        // 2M
    const long long NW = (long long)G4 * EMBc;        // 1M per matrix
    long long gid = (long long)blockIdx.x * blockDim.x + threadIdx.x;
    if (gid < NE) {
        long long row = gid >> 9;                     // /512
        long long col = gid & 511;
        int tok = sent[row];
        emb16[gid] = f32_to_bf16_rne(wembed[(long long)tok * EMBc + col]);
        return;
    }
    gid -= NE;
    if (gid < 2 * NW) {
        float v = (gid < NW) ? wihf[gid] : wihb[gid - NW];
        wih16[gid] = f32_to_bf16_rne(v);
        return;
    }
    gid -= 2 * NW;
    if (gid < 2 * NW) {
        float v = (gid < NW) ? whhf[gid] : whhb[gid - NW];
        whh16[gid] = f32_to_bf16_rne(v);
    }
}

// ---------------- K1: xproj[d][L][4H] = emb @ W_ih_d^T + b_d  (WMMA GEMM) ----------------
// grid: (8, 256, 2)  block: 128 (4 waves).  Each wave: 4 N-tiles sharing A fragments.
// Software-pipelined over k-steps: B fragments of step kk+1 are loaded (double buffer) before
// the WMMAs of step kk issue, so loads overlap the XDL pipe within a single wave.
__global__ __launch_bounds__(128) void gemm_xproj(const unsigned short* __restrict__ emb16,
                                                  const unsigned short* __restrict__ wih16,
                                                  const float* __restrict__ bias_f,
                                                  const float* __restrict__ bias_b,
                                                  float* __restrict__ xproj) {
    const int d    = blockIdx.z;
    const unsigned short* Wd = wih16 + (size_t)d * G4 * EMBc;
    const float* bias = d ? bias_b : bias_f;
    float* XP = xproj + (size_t)d * Lc * G4;

    const int wave = threadIdx.x >> 5;
    const int lane = threadIdx.x & 31;
    const int half = lane >> 4;       // 0: lanes 0-15, 1: lanes 16-31
    const int ln   = lane & 15;

    const int m0    = blockIdx.y * 16;
    const int nbase = blockIdx.x * 256 + wave * 64 + ln;   // this lane's B column, tile ti adds ti*16
    const int row   = m0 + ln;                              // this lane's A row

    const unsigned short* Arow = emb16 + (size_t)row * EMBc;
    const unsigned short* B0 = Wd + (size_t)(nbase)      * EMBc;
    const unsigned short* B1 = Wd + (size_t)(nbase + 16) * EMBc;
    const unsigned short* B2 = Wd + (size_t)(nbase + 32) * EMBc;
    const unsigned short* B3 = Wd + (size_t)(nbase + 48) * EMBc;

    v8f acc[4];
    #pragma unroll
    for (int ti = 0; ti < 4; ++ti)
        #pragma unroll
        for (int i = 0; i < 8; ++i) acc[ti][i] = 0.0f;

    BFrags cur, nxt;
    load_b4(cur, B0, B1, B2, B3, 0, half);

    #pragma unroll
    for (int kk = 0; kk < 16; ++kk) {
        const int k0 = kk * 32;
        // A 16-bit 16x32: lane half 0 -> K 0..7 & 16..23 ; half 1 -> K 8..15 & 24..31
        Frag16 fa;
        fa.h[0] = *(const v8bf*)(Arow + k0 + half * 8);
        fa.h[1] = *(const v8bf*)(Arow + k0 + half * 8 + 16);
        if (kk < 15) load_b4(nxt, B0, B1, B2, B3, k0 + 32, half);   // prefetch next k-step
        acc[0] = __builtin_amdgcn_wmma_f32_16x16x32_bf16(false, fa.v, false, cur.b0.v, (short)0, acc[0], false, false);
        acc[1] = __builtin_amdgcn_wmma_f32_16x16x32_bf16(false, fa.v, false, cur.b1.v, (short)0, acc[1], false, false);
        acc[2] = __builtin_amdgcn_wmma_f32_16x16x32_bf16(false, fa.v, false, cur.b2.v, (short)0, acc[2], false, false);
        acc[3] = __builtin_amdgcn_wmma_f32_16x16x32_bf16(false, fa.v, false, cur.b3.v, (short)0, acc[3], false, false);
        if (kk < 15) cur = nxt;                                     // SSA rename after full unroll
    }
    // D layout: lane l, VGPR v -> (M = v + 8*half, N = ln)
    #pragma unroll
    for (int ti = 0; ti < 4; ++ti) {
        const int n = nbase + ti * 16;
        const float b = bias[n];
        #pragma unroll
        for (int v = 0; v < 8; ++v) {
            const int M = v + half * 8;
            XP[(size_t)(m0 + M) * G4 + n] = acc[ti][v] + b;
        }
    }
}

// ---------------- K2: sequential LSTM scan, one workgroup per direction ----------------
// Recurrent matvec via broadcast-WMMA: A = h replicated across all 16 rows (two ds_load_b128 from
// packed-bf16 h in LDS, shared across the wave's 4 tiles), B = W_hh rows streamed from L2 with a
// software-pipelined double buffer, C = xproj[t] row rides in as the accumulator init.
// Next step's xproj row is prefetched from HBM during compute.
__global__ __launch_bounds__(1024) void lstm_scan(const unsigned short* __restrict__ whh16,
                                                  const float* __restrict__ xproj,
                                                  float* __restrict__ hout) {
    const int d = blockIdx.x;
    const unsigned short* W  = whh16 + (size_t)d * G4 * Hc;
    const float*          XP = xproj + (size_t)d * Lc * G4;
    float*                HO = hout  + (size_t)d * Lc * Hc;

    __shared__ unsigned short h_pk[Hc];   // h in bf16, packed for direct A-fragment b128 loads
    __shared__ float gates[G4];           // pre-activation gates for this step

    const int tid  = threadIdx.x;
    const int lane = tid & 31;
    const int wave = tid >> 5;
    const int half = lane >> 4;
    const int ln   = lane & 15;

    if (tid < Hc) h_pk[tid] = 0;          // h0 = 0
    float creg = 0.0f;                    // c for element `tid` (threads 0..511)
    __syncthreads();

    const int nbase = wave * 64 + ln;     // this lane's base gate column; tile ti adds ti*16
    const unsigned short* B0 = W + (size_t)(nbase)      * Hc;
    const unsigned short* B1 = W + (size_t)(nbase + 16) * Hc;
    const unsigned short* B2 = W + (size_t)(nbase + 32) * Hc;
    const unsigned short* B3 = W + (size_t)(nbase + 48) * Hc;

    for (int t = 0; t < Lc; ++t) {
        const int rt = d ? (Lc - 1 - t) : t;
        const float* xrow = XP + (size_t)rt * G4;

        // hide HBM latency of the only cold stream: prefetch next step's xproj row (8 KB)
        if (t + 1 < Lc && tid < 64) {
            const int rtn = d ? (Lc - 2 - t) : (t + 1);
            __builtin_prefetch(XP + (size_t)rtn * G4 + tid * 32, 0, 0);
        }

        // ---- gates[n] = xproj[rt][n] + dot(h, W_hh[n])  via broadcast-WMMA ----
        v8f acc[4];
        #pragma unroll
        for (int ti = 0; ti < 4; ++ti) {
            const float xv = xrow[nbase + ti * 16];     // C column value (same for all M rows)
            #pragma unroll
            for (int i = 0; i < 8; ++i) acc[ti][i] = xv;
        }
        BFrags cur, nxt;
        load_b4(cur, B0, B1, B2, B3, 0, half);
        #pragma unroll
        for (int kk = 0; kk < 16; ++kk) {
            const int k0 = kk * 32;
            Frag16 fa;                                   // shared across the 4 tiles (LDS, low latency)
            fa.h[0] = *(const v8bf*)(h_pk + k0 + half * 8);        // ds_load_b128
            fa.h[1] = *(const v8bf*)(h_pk + k0 + half * 8 + 16);   // ds_load_b128
            if (kk < 15) load_b4(nxt, B0, B1, B2, B3, k0 + 32, half);
            acc[0] = __builtin_amdgcn_wmma_f32_16x16x32_bf16(false, fa.v, false, cur.b0.v, (short)0, acc[0], false, false);
            acc[1] = __builtin_amdgcn_wmma_f32_16x16x32_bf16(false, fa.v, false, cur.b1.v, (short)0, acc[1], false, false);
            acc[2] = __builtin_amdgcn_wmma_f32_16x16x32_bf16(false, fa.v, false, cur.b2.v, (short)0, acc[2], false, false);
            acc[3] = __builtin_amdgcn_wmma_f32_16x16x32_bf16(false, fa.v, false, cur.b3.v, (short)0, acc[3], false, false);
            if (kk < 15) cur = nxt;
        }
        if (lane < 16) {
            #pragma unroll
            for (int ti = 0; ti < 4; ++ti) gates[nbase + ti * 16] = acc[ti][0];  // M=0 row
        }
        __syncthreads();

        // ---- pointwise LSTM cell update (PyTorch gate order i,f,g,o) ----
        if (tid < Hc) {
            const int j = tid;
            const float i  = fsigmoid(gates[j]);
            const float f  = fsigmoid(gates[Hc + j]);
            const float g  = ftanh   (gates[2 * Hc + j]);
            const float o  = fsigmoid(gates[3 * Hc + j]);
            creg = f * creg + i * g;
            const float hn = o * ftanh(creg);
            HO[(size_t)rt * Hc + j] = hn;
            h_pk[j] = f32_to_bf16_rne(hn);
        }
        __syncthreads();
    }
}

// ---------------- K3: feats[L][8] = concat(hf,hb) @ w_tag^T + b_tag ----------------
__global__ void tag_proj(const float* __restrict__ hf, const float* __restrict__ hb,
                         const float* __restrict__ wtag, const float* __restrict__ btag,
                         float* __restrict__ feats) {
    int gid = blockIdx.x * blockDim.x + threadIdx.x;
    if (gid >= Lc * TAGSc) return;
    const int row = gid >> 3;
    const int tag = gid & 7;
    const float* wt = wtag + (size_t)tag * (2 * Hc);
    const float* a  = hf + (size_t)row * Hc;
    const float* b  = hb + (size_t)row * Hc;
    float s = btag[tag];
    for (int k = 0; k < Hc; ++k) s += a[k] * wt[k];
    for (int k = 0; k < Hc; ++k) s += b[k] * wt[Hc + k];
    feats[gid] = s;
}

// ---------------- K4: Viterbi forward recurrence + backtrack (single wave) ----------------
__global__ void viterbi(const float* __restrict__ feats, const float* __restrict__ trans,
                        int* __restrict__ bptrs, float* __restrict__ out, int out_size) {
    __shared__ float fvs[TAGSc];
    __shared__ float term[TAGSc];
    const int lane = threadIdx.x;

    float tr[TAGSc];
    if (lane < TAGSc) {
        #pragma unroll
        for (int p = 0; p < TAGSc; ++p) tr[p] = trans[lane * TAGSc + p];
        fvs[lane] = (lane == STARTc) ? 0.0f : NEGc;
    }
    __syncthreads();

    for (int t = 0; t < Lc; ++t) {
        float best = NEGc * 4.0f;
        int bp = 0;
        if (lane < TAGSc) {
            best = fvs[0] + tr[0];
            #pragma unroll
            for (int p = 1; p < TAGSc; ++p) {
                float s = fvs[p] + tr[p];
                if (s > best) { best = s; bp = p; }
            }
        }
        __syncthreads();
        if (lane < TAGSc) {
            fvs[lane] = best + feats[t * TAGSc + lane];
            bptrs[t * TAGSc + lane] = bp;
        }
        __syncthreads();
    }

    if (lane < TAGSc) term[lane] = fvs[lane] + trans[STOPc * TAGSc + lane];
    __syncthreads();

    if (lane == 0) {
        float best = term[0];
        int bi = 0;
        #pragma unroll
        for (int n = 1; n < TAGSc; ++n)
            if (term[n] > best) { best = term[n]; bi = n; }
        if (out_size > 0) out[0] = best;
        int cur = bi;
        for (int t = Lc - 1; t >= 0; --t) {       // path[L-1]=best ; path[t-1]=bptrs[t][path[t]]
            if (1 + t < out_size) out[1 + t] = (float)cur;
            cur = bptrs[t * TAGSc + cur];
        }
    }
}

// ---------------- host side ----------------
extern "C" void kernel_launch(void* const* d_in, const int* in_sizes, int n_in,
                              void* d_out, int out_size, void* d_ws, size_t ws_size,
                              hipStream_t stream) {
    const int*   sent   = (const int*)  d_in[0];
    const float* wembed = (const float*)d_in[1];
    const float* wihf   = (const float*)d_in[2];
    const float* whhf   = (const float*)d_in[3];
    const float* bf     = (const float*)d_in[4];
    const float* wihb   = (const float*)d_in[5];
    const float* whhb   = (const float*)d_in[6];
    const float* bb     = (const float*)d_in[7];
    const float* wtag   = (const float*)d_in[8];
    const float* btag   = (const float*)d_in[9];
    const float* trans  = (const float*)d_in[10];
    float* out = (float*)d_out;

    // workspace carve-up (256B aligned)
    char* ws = (char*)d_ws;
    size_t off = 0;
    auto carve = [&](size_t bytes) -> char* {
        char* p = ws + off;
        off = (off + bytes + 255) & ~(size_t)255;
        return p;
    };
    unsigned short* emb16 = (unsigned short*)carve((size_t)Lc * EMBc * 2);        // 4 MiB
    unsigned short* wih16 = (unsigned short*)carve((size_t)2 * G4 * EMBc * 2);    // 4 MiB
    unsigned short* whh16 = (unsigned short*)carve((size_t)2 * G4 * Hc * 2);      // 4 MiB
    float* xproj = (float*)carve((size_t)2 * Lc * G4 * 4);                        // 64 MiB
    float* hout  = (float*)carve((size_t)2 * Lc * Hc * 4);                        // 16 MiB
    float* feats = (float*)carve((size_t)Lc * TAGSc * 4);
    int*   bptrs = (int*)  carve((size_t)Lc * TAGSc * 4);
    (void)ws_size; (void)in_sizes; (void)n_in;

    // K0: gather + bf16 conversion
    const long long total0 = (long long)Lc * EMBc + 4LL * G4 * EMBc;              // 6,291,456
    prep_kernel<<<(int)((total0 + 255) / 256), 256, 0, stream>>>(
        sent, wembed, wihf, wihb, whhf, whhb, emb16, wih16, whh16);

    // K1: input projections (both directions), WMMA GEMM
    gemm_xproj<<<dim3(8, 256, 2), 128, 0, stream>>>(emb16, wih16, bf, bb, xproj);

    // K2: sequential LSTM scans (block 0 = forward, block 1 = backward)
    lstm_scan<<<2, 1024, 0, stream>>>(whh16, xproj, hout);

    // K3: tag-space projection
    tag_proj<<<(Lc * TAGSc + 255) / 256, 256, 0, stream>>>(
        hout, hout + (size_t)Lc * Hc, wtag, btag, feats);

    // K4: Viterbi + backtrack
    viterbi<<<1, 32, 0, stream>>>(feats, trans, bptrs, out, out_size);
}
